// HeteroGNN_58291296141322
// MI455X (gfx1250) — matrix operand order repaired and checked by reading, not compiled, not run on hardware
//
#include <hip/hip_runtime.h>

// ---------- types ----------
typedef __bf16 bf16;
typedef __attribute__((ext_vector_type(16))) __bf16 v16bf;
typedef __attribute__((ext_vector_type(8)))  __bf16 v8bf;
typedef __attribute__((ext_vector_type(4)))  __bf16 v4bf;
typedef __attribute__((ext_vector_type(8)))  float  v8f;

#define C_DIM 512
#define WAVES_PER_BLOCK 5
#define GEMM_THREADS (WAVES_PER_BLOCK * 32)
#define KQ 128            // k-quarter staged per pipeline stage
#define LDS_STRIDE 136    // 128 + 8 pad: row stride 272B = 68 dwords -> +4 banks/row
#define BUF_ELEMS (64 * LDS_STRIDE)
#define BUF_BYTES (BUF_ELEMS * 2)
#define FILL_ITERS 7      // ceil(64*16 / 160): every wave issues exactly 7 async loads
#define N_STAGES 12       // 3 matrices x 4 k-quarters

// ---------- utility kernels ----------
__global__ void k_zero_f32(float* __restrict__ p, long n) {
  long i = (long)blockIdx.x * blockDim.x + threadIdx.x;
  long v = i * 4;
  if (v >= n) return;
  if (v + 3 < n) {
    *(float4*)(p + v) = make_float4(0.f, 0.f, 0.f, 0.f);
  } else {
    for (long j = v; j < n; ++j) p[j] = 0.f;
  }
}

__global__ void k_f32_to_bf16(const float* __restrict__ in, bf16* __restrict__ out, long n) {
  long i = (long)blockIdx.x * blockDim.x + threadIdx.x;
  if (i < n) out[i] = (bf16)in[i];
}

// Wrc[l,e,:,:] = bf16( Wr[l,e,0,:,:] + Wr[l,e,1,:,:] )
__global__ void k_make_wrc(const float* __restrict__ Wr, bf16* __restrict__ Wrc, long n, long CC) {
  long i = (long)blockIdx.x * blockDim.x + threadIdx.x;
  if (i >= n) return;
  long le = i / CC, r = i - le * CC;
  Wrc[i] = (bf16)(Wr[(le * 2 + 0) * CC + r] + Wr[(le * 2 + 1) * CC + r]);
}

// bsum[l,e,c] = bl[l,e,0,c] + bl[l,e,1,c]
__global__ void k_make_bsum(const float* __restrict__ bl, float* __restrict__ bsum, int n) {
  int i = blockIdx.x * blockDim.x + threadIdx.x;
  if (i >= n) return;
  int le = i / C_DIM, c = i - le * C_DIM;
  bsum[i] = bl[(le * 2 + 0) * C_DIM + c] + bl[(le * 2 + 1) * C_DIM + c];
}

// ---------- edge scatter (segment_sum + count) ----------
__global__ __launch_bounds__(128)
void k_scatter(const bf16* __restrict__ xsrc, const int* __restrict__ esrc,
               const int* __restrict__ edst, float* __restrict__ s,
               float* __restrict__ cnt) {
  int e = blockIdx.x;
  int src = esrc[e];
  int dst = edst[e];
  int t = threadIdx.x;
  v4bf x4 = *(const v4bf*)(xsrc + (long)src * C_DIM + t * 4);
  float* d = s + (long)dst * C_DIM + t * 4;
  atomicAdd(d + 0, (float)x4[0]);
  atomicAdd(d + 1, (float)x4[1]);
  atomicAdd(d + 2, (float)x4[2]);
  atomicAdd(d + 3, (float)x4[3]);
  if (t == 0) atomicAdd(cnt + dst, 1.0f);
}

// s_bf = bf16(s); m_bf = bf16(s / max(cnt,1))
__global__ void k_finalize(const float* __restrict__ s, const float* __restrict__ cnt,
                           bf16* __restrict__ s_bf, bf16* __restrict__ m_bf, long n_rows) {
  long i = (long)blockIdx.x * blockDim.x + threadIdx.x;
  long v = i * 4;
  if (v >= n_rows * C_DIM) return;
  long row = v >> 9;
  float inv = 1.0f / fmaxf(cnt[row], 1.0f);
  float4 sv = *(const float4*)(s + v);
  v4bf sb, mb;
  sb[0] = (bf16)sv.x; sb[1] = (bf16)sv.y; sb[2] = (bf16)sv.z; sb[3] = (bf16)sv.w;
  mb[0] = (bf16)(sv.x * inv); mb[1] = (bf16)(sv.y * inv);
  mb[2] = (bf16)(sv.z * inv); mb[3] = (bf16)(sv.w * inv);
  *(v4bf*)(s_bf + v) = sb;
  *(v4bf*)(m_bf + v) = mb;
}

// ---------- fused triple-GEMM via WMMA, double-buffered async-LDS weights ----------
// out[n,512] = A0@W0^T + A1@W1^T + A2@W2^T + bias
// 5 waves/block; wave w owns a 16-row strip x 64 cols (4 accumulators).
// Pipeline over 12 stages (matrix mi = s>>2, k-quarter kq = s&3):
//   issue async fill(s+1, buf^1) -> s_wait_asynccnt 7 (fill(s) done, fill(s+1)
//   still in flight) -> barrier -> WMMA on buf -> barrier.
__global__ __launch_bounds__(GEMM_THREADS)
void k_gemm3_wmma(const bf16* __restrict__ A0, const bf16* __restrict__ A1,
                  const bf16* __restrict__ A2,
                  const bf16* __restrict__ W0, const bf16* __restrict__ W1,
                  const bf16* __restrict__ W2,
                  const float* __restrict__ bias, float* __restrict__ out,
                  int n_strips) {
  __shared__ __align__(16) bf16 lds_w[2 * BUF_ELEMS];  // 2 x ~17 KB

  const int tid = threadIdx.x;
  const int wave = tid >> 5;
  const int lane = tid & 31;
  const int lo = lane & 15;
  const int hi = lane >> 4;
  const int strip = blockIdx.x * WAVES_PER_BLOCK + wave;
  const bool active = (strip < n_strips);  // wave-uniform
  const long rowA = (long)(strip * 16 + lo) * C_DIM;
  const int colbase = blockIdx.y * 64;
  const unsigned lds_base = (unsigned)(uintptr_t)lds_w;

  v8f acc0 = {}, acc1 = {}, acc2 = {}, acc3 = {};

  // Exactly FILL_ITERS async-load issues per wave (clamped tail duplicates the
  // last chunk, writing identical data twice -> harmless).
#define FILL(WPTR, S, BUF)                                                     \
  do {                                                                         \
    const int kb_ = ((S) & 3) * KQ;                                            \
    for (int it_ = 0; it_ < FILL_ITERS; ++it_) {                               \
      int idx_ = tid + it_ * GEMM_THREADS;                                     \
      idx_ = idx_ > 1023 ? 1023 : idx_;                                        \
      int row_ = idx_ >> 4;                                                    \
      int ch_ = idx_ & 15;                                                     \
      const bf16* g_ = (WPTR) + (long)(colbase + row_) * C_DIM + kb_ + ch_ * 8;\
      unsigned laddr_ = lds_base + (unsigned)(BUF) * BUF_BYTES +               \
                        2u * (unsigned)(row_ * LDS_STRIDE + ch_ * 8);          \
      asm volatile("global_load_async_to_lds_b128 %0, %1, off"                 \
                   :: "v"(laddr_), "v"(g_) : "memory");                        \
    }                                                                          \
  } while (0)

#define COMPUTE(APTR, S, BUF)                                                  \
  do {                                                                         \
    if (active) { /* wave-uniform: EXEC all-ones for WMMA */                   \
      const int kb_ = ((S) & 3) * KQ;                                          \
      const bf16* lw_ = lds_w + (BUF) * BUF_ELEMS;                             \
      _Pragma("unroll") for (int k0 = 0; k0 < KQ; k0 += 32) {                  \
        const long ab_ = rowA + kb_ + k0 + hi * 16;                            \
        v8bf alo_ = *(const v8bf*)((APTR) + ab_);                              \
        v8bf ahi_ = *(const v8bf*)((APTR) + ab_ + 8);                          \
        v16bf a_ = __builtin_shufflevector(alo_, ahi_, 0, 1, 2, 3, 4, 5, 6, 7, \
                                           8, 9, 10, 11, 12, 13, 14, 15);      \
        const int wk_ = k0 + hi * 16;                                          \
        v16bf b0_ = __builtin_shufflevector(                                   \
            *(const v8bf*)&lw_[(0 + lo) * LDS_STRIDE + wk_],                   \
            *(const v8bf*)&lw_[(0 + lo) * LDS_STRIDE + wk_ + 8],               \
            0, 1, 2, 3, 4, 5, 6, 7, 8, 9, 10, 11, 12, 13, 14, 15);             \
        v16bf b1_ = __builtin_shufflevector(                                   \
            *(const v8bf*)&lw_[(16 + lo) * LDS_STRIDE + wk_],                  \
            *(const v8bf*)&lw_[(16 + lo) * LDS_STRIDE + wk_ + 8],              \
            0, 1, 2, 3, 4, 5, 6, 7, 8, 9, 10, 11, 12, 13, 14, 15);             \
        v16bf b2_ = __builtin_shufflevector(                                   \
            *(const v8bf*)&lw_[(32 + lo) * LDS_STRIDE + wk_],                  \
            *(const v8bf*)&lw_[(32 + lo) * LDS_STRIDE + wk_ + 8],              \
            0, 1, 2, 3, 4, 5, 6, 7, 8, 9, 10, 11, 12, 13, 14, 15);             \
        v16bf b3_ = __builtin_shufflevector(                                   \
            *(const v8bf*)&lw_[(48 + lo) * LDS_STRIDE + wk_],                  \
            *(const v8bf*)&lw_[(48 + lo) * LDS_STRIDE + wk_ + 8],              \
            0, 1, 2, 3, 4, 5, 6, 7, 8, 9, 10, 11, 12, 13, 14, 15);             \
        acc0 = __builtin_amdgcn_wmma_f32_16x16x32_bf16(false, a_, false, b0_, (short)0, acc0, false, false); \
        acc1 = __builtin_amdgcn_wmma_f32_16x16x32_bf16(false, a_, false, b1_, (short)0, acc1, false, false); \
        acc2 = __builtin_amdgcn_wmma_f32_16x16x32_bf16(false, a_, false, b2_, (short)0, acc2, false, false); \
        acc3 = __builtin_amdgcn_wmma_f32_16x16x32_bf16(false, a_, false, b3_, (short)0, acc3, false, false); \
      }                                                                        \
    }                                                                          \
  } while (0)

  FILL(W0, 0, 0);  // prologue: stage 0 fill into buffer 0

  for (int s = 0; s < N_STAGES; ++s) {
    const int buf = s & 1;
    if (s + 1 < N_STAGES) {
      const int ns = s + 1;
      const int nbuf = ns & 1;
      switch (ns >> 2) {  // direct kernel-arg pointers -> global_load path
        case 0:  FILL(W0, ns, nbuf); break;
        case 1:  FILL(W1, ns, nbuf); break;
        default: FILL(W2, ns, nbuf); break;
      }
      // fill(s) complete while the 7 just-issued fill(s+1) ops stay in flight
      asm volatile("s_wait_asynccnt 0x7" ::: "memory");
    } else {
      asm volatile("s_wait_asynccnt 0x0" ::: "memory");
    }
    __syncthreads();
    switch (s >> 2) {
      case 0:  COMPUTE(A0, s, buf); break;
      case 1:  COMPUTE(A1, s, buf); break;
      default: COMPUTE(A2, s, buf); break;
    }
    __syncthreads();  // reads done before buf is refilled two stages later
  }
#undef FILL
#undef COMPUTE

  if (active) {
    // D layout: VGPR v -> row = strip*16 + v + hi*8 ; col = colbase + ct*16 + lo
#define TILE_STORE(ACC, CT)                                                     \
  {                                                                             \
    int col = colbase + (CT) * 16 + lo;                                         \
    float bv = bias[col];                                                       \
    _Pragma("unroll") for (int v = 0; v < 8; ++v) {                             \
      out[(long)(strip * 16 + v + hi * 8) * C_DIM + col] = (ACC)[v] + bv;       \
    }                                                                           \
  }
    TILE_STORE(acc0, 0)
    TILE_STORE(acc1, 1)
    TILE_STORE(acc2, 2)
    TILE_STORE(acc3, 3)
#undef TILE_STORE
  }
}

// ---------- LayerNorm + ReLU (one block per row) ----------
__global__ __launch_bounds__(256)
void k_ln_relu(const float* __restrict__ in, const float* __restrict__ g,
               const float* __restrict__ b, float* __restrict__ out_f32,
               bf16* __restrict__ out_bf) {
  __shared__ float s_sum[256];
  __shared__ float s_sq[256];
  const int row = blockIdx.x;
  const int t = threadIdx.x;
  const float* x = in + (long)row * C_DIM;
  float x0 = x[t];
  float x1 = x[t + 256];
  s_sum[t] = x0 + x1;
  s_sq[t] = x0 * x0 + x1 * x1;
  __syncthreads();
  for (int off = 128; off > 0; off >>= 1) {
    if (t < off) {
      s_sum[t] += s_sum[t + off];
      s_sq[t] += s_sq[t + off];
    }
    __syncthreads();
  }
  const float mu = s_sum[0] * (1.0f / 512.0f);
  const float var = s_sq[0] * (1.0f / 512.0f) - mu * mu;
  const float inv = rsqrtf(var + 1e-5f);
  float y0 = fmaxf(g[t] * (x0 - mu) * inv + b[t], 0.0f);
  float y1 = fmaxf(g[t + 256] * (x1 - mu) * inv + b[t + 256], 0.0f);
  if (out_f32) {
    float* o = out_f32 + (long)row * C_DIM;
    o[t] = y0;
    o[t + 256] = y1;
  }
  if (out_bf) {
    bf16* o = out_bf + (long)row * C_DIM;
    o[t] = (bf16)y0;
    o[t + 256] = (bf16)y1;
  }
}

// ---------- host launcher ----------
extern "C" void kernel_launch(void* const* d_in, const int* in_sizes, int n_in,
                              void* d_out, int out_size, void* d_ws, size_t ws_size,
                              hipStream_t stream) {
  const float* x_user = (const float*)d_in[0];
  const float* x_item = (const float*)d_in[1];
  const float* Wl     = (const float*)d_in[2];
  const float* bl     = (const float*)d_in[3];
  const float* Wr     = (const float*)d_in[4];
  const float* ln_g   = (const float*)d_in[5];
  const float* ln_b   = (const float*)d_in[6];
  const int*   edge_ui = (const int*)d_in[7];
  const int*   edge_iu = (const int*)d_in[8];

  const int C = C_DIM;
  const int Nu = in_sizes[0] / C;
  const int Ni = in_sizes[1] / C;
  const int E_ui = in_sizes[7] / 2;
  const int E_iu = in_sizes[8] / 2;
  const int L = in_sizes[5] / (2 * C);  // ln_g [L,2,C]
  const long CC = (long)C * C;
  const int Nmax = (Nu > Ni) ? Nu : Ni;

  // workspace carve (256B aligned)
  char* p = (char*)d_ws;
  auto carve = [&](size_t bytes) -> char* {
    char* r = p;
    p += (bytes + 255) & ~(size_t)255;
    return r;
  };
  float* s_buf  = (float*)carve((size_t)Nmax * C * 4);
  float* cnt    = (float*)carve((size_t)Nmax * 4);
  bf16*  xu_bf  = (bf16*)carve((size_t)Nu * C * 2);
  bf16*  xi_bf  = (bf16*)carve((size_t)Ni * C * 2);
  bf16*  s_bf   = (bf16*)carve((size_t)Nmax * C * 2);
  bf16*  m_bf   = (bf16*)carve((size_t)Nmax * C * 2);
  bf16*  Wl_bf  = (bf16*)carve((size_t)L * 4 * CC * 2);
  bf16*  Wrc_bf = (bf16*)carve((size_t)L * 2 * CC * 2);
  float* bsum   = (float*)carve((size_t)L * 2 * C * 4);

  // pre-LN activations live in d_out: [pre_u ; pre_i]
  float* pre_u = (float*)d_out;
  float* pre_i = (float*)d_out + (long)Nu * C;

  // ---- one-time weight / input prep ----
  {
    long nWl = (long)L * 4 * CC;
    k_f32_to_bf16<<<(unsigned)((nWl + 255) / 256), 256, 0, stream>>>(Wl, Wl_bf, nWl);
    long nWrc = (long)L * 2 * CC;
    k_make_wrc<<<(unsigned)((nWrc + 255) / 256), 256, 0, stream>>>(Wr, Wrc_bf, nWrc, CC);
    int nb = L * 2 * C;
    k_make_bsum<<<(unsigned)((nb + 255) / 256), 256, 0, stream>>>(bl, bsum, nb);
    long nxu = (long)Nu * C;
    k_f32_to_bf16<<<(unsigned)((nxu + 255) / 256), 256, 0, stream>>>(x_user, xu_bf, nxu);
    long nxi = (long)Ni * C;
    k_f32_to_bf16<<<(unsigned)((nxi + 255) / 256), 256, 0, stream>>>(x_item, xi_bf, nxi);
  }

  for (int l = 0; l < L; ++l) {
    // ---- direction 0: user -> item (edge type 0), produces pre_i ----
    {
      long ns = (long)Ni * C;
      k_zero_f32<<<(unsigned)((ns / 4 + 255) / 256), 256, 0, stream>>>(s_buf, ns);
      k_zero_f32<<<(unsigned)(((Ni + 3) / 4 + 255) / 256), 256, 0, stream>>>(cnt, Ni);
      k_scatter<<<E_ui, 128, 0, stream>>>(xu_bf, edge_ui, edge_ui + E_ui, s_buf, cnt);
      k_finalize<<<(unsigned)((ns / 4 + 255) / 256), 256, 0, stream>>>(s_buf, cnt, s_bf, m_bf, Ni);
      const bf16* W0 = Wl_bf + ((long)(l * 2 + 0) * 2 + 0) * CC;
      const bf16* W1 = Wl_bf + ((long)(l * 2 + 0) * 2 + 1) * CC;
      const bf16* W2 = Wrc_bf + (long)(l * 2 + 0) * CC;
      const float* bs = bsum + (l * 2 + 0) * C;
      int n_strips = Ni / 16;
      dim3 grid((n_strips + WAVES_PER_BLOCK - 1) / WAVES_PER_BLOCK, C / 64);
      k_gemm3_wmma<<<grid, GEMM_THREADS, 0, stream>>>(s_bf, m_bf, xi_bf, W0, W1, W2, bs, pre_i, n_strips);
    }
    // ---- direction 1: item -> user (edge type 1), produces pre_u ----
    {
      long ns = (long)Nu * C;
      k_zero_f32<<<(unsigned)((ns / 4 + 255) / 256), 256, 0, stream>>>(s_buf, ns);
      k_zero_f32<<<(unsigned)(((Nu + 3) / 4 + 255) / 256), 256, 0, stream>>>(cnt, Nu);
      k_scatter<<<E_iu, 128, 0, stream>>>(xi_bf, edge_iu, edge_iu + E_iu, s_buf, cnt);
      k_finalize<<<(unsigned)((ns / 4 + 255) / 256), 256, 0, stream>>>(s_buf, cnt, s_bf, m_bf, Nu);
      const bf16* W0 = Wl_bf + ((long)(l * 2 + 1) * 2 + 0) * CC;
      const bf16* W1 = Wl_bf + ((long)(l * 2 + 1) * 2 + 1) * CC;
      const bf16* W2 = Wrc_bf + (long)(l * 2 + 1) * CC;
      const float* bs = bsum + (l * 2 + 1) * C;
      int n_strips = Nu / 16;
      dim3 grid((n_strips + WAVES_PER_BLOCK - 1) / WAVES_PER_BLOCK, C / 64);
      k_gemm3_wmma<<<grid, GEMM_THREADS, 0, stream>>>(s_bf, m_bf, xu_bf, W0, W1, W2, bs, pre_u, n_strips);
    }
    // ---- per-node-type LayerNorm + ReLU ----
    const bool last = (l == L - 1);
    k_ln_relu<<<Nu, 256, 0, stream>>>(pre_u, ln_g + (l * 2 + 0) * C, ln_b + (l * 2 + 0) * C,
                                      last ? pre_u : nullptr, last ? nullptr : xu_bf);
    k_ln_relu<<<Ni, 256, 0, stream>>>(pre_i, ln_g + (l * 2 + 1) * C, ln_b + (l * 2 + 1) * C,
                                      last ? pre_i : nullptr, last ? nullptr : xi_bf);
  }
}
